// Block_4776003633552
// MI455X (gfx1250) — matrix operand order repaired and checked by reading, not compiled, not run on hardware
//
#include <hip/hip_runtime.h>
#include <hip/hip_bf16.h>

#define B_ 4
#define T_ 2048
#define C_ 512
#define H_ 8
#define D_ 64
#define EPS_ 1e-5f

typedef __attribute__((ext_vector_type(16))) __bf16 v16bf;
typedef __attribute__((ext_vector_type(8)))  __bf16 v8bf;
typedef __attribute__((ext_vector_type(8)))  float  v8f;

static __device__ __forceinline__ __bf16 f2bf(float f) {
  unsigned int u = __builtin_bit_cast(unsigned int, f);
  unsigned int r = u + 0x7fffu + ((u >> 16) & 1u);   // round-to-nearest-even
  unsigned short h = (unsigned short)(r >> 16);
  return __builtin_bit_cast(__bf16, h);
}

static __device__ __forceinline__ v16bf cat8(v8bf lo, v8bf hi) {
  return __builtin_shufflevector(lo, hi, 0, 1, 2, 3, 4, 5, 6, 7,
                                 8, 9, 10, 11, 12, 13, 14, 15);
}

// ---- WMMA fragment loaders (CDNA5 ISA 7.12.2, 16-bit, wave32) ----
// A (16x32, MxK), row-major source: lane%16=m, khi=lane/16.
//   e<8: k = e + 8*khi ; e>=8: k = 16 + (e-8) + 8*khi  -> two 16B runs.
static __device__ __forceinline__ v16bf load_a(const __bf16* A, int lda, int row0,
                                               int k0, int lane) {
  int m = lane & 15, khi = lane >> 4;
  const __bf16* p = A + (size_t)(row0 + m) * lda + k0 + 8 * khi;
  v8bf lo = *(const v8bf*)p;
  v8bf hi = *(const v8bf*)(p + 16);
  return cat8(lo, hi);
}

// B (32x16, KxN) from a [N x K] row-major (K-contiguous) source:
//   lane%16=n, khi=lane/16, e<8: k = e + 8*khi ; e>=8: k = 16 + (e-8) + 8*khi -> two 16B runs.
static __device__ __forceinline__ v16bf load_bt(const __bf16* Mat, int ldm, int n0,
                                                int k0, int lane) {
  int n = lane & 15, khi = lane >> 4;
  const __bf16* p = Mat + (size_t)(n0 + n) * ldm + k0 + 8 * khi;
  v8bf lo = *(const v8bf*)p;
  v8bf hi = *(const v8bf*)(p + 16);
  return cat8(lo, hi);
}

static __device__ __forceinline__ v8f wmma_bf16(v16bf a, v16bf b, v8f c) {
  return __builtin_amdgcn_wmma_f32_16x16x32_bf16(false, a, false, b, (short)0, c,
                                                 false, false);
}

// -------------------------------------------------------------------------
// Convert f32 [batch x K x N] -> bf16 transposed [batch x N x K].
__global__ void cvt_t_kernel(const float* __restrict__ s, __bf16* __restrict__ d,
                             int K, int N, int total) {
  int i = blockIdx.x * blockDim.x + threadIdx.x;
  if (i >= total) return;
  int kn = K * N;
  int b = i / kn, r = i % kn;
  int kk = r / N, nn = r % N;
  d[(size_t)b * kn + (size_t)nn * K + kk] = f2bf(s[i]);
}

// LayerNorm over C=512; one 256-thread block per row; bf16 output.
__global__ void ln_kernel(const float* __restrict__ in, const float* __restrict__ g,
                          const float* __restrict__ be, __bf16* __restrict__ out) {
  int row = blockIdx.x, tid = threadIdx.x;
  const float* r = in + (size_t)row * C_;
  float x0 = r[tid], x1 = r[tid + 256];
  __shared__ float s1[256], s2[256];
  s1[tid] = x0 + x1;
  s2[tid] = x0 * x0 + x1 * x1;
  __syncthreads();
  for (int o = 128; o > 0; o >>= 1) {
    if (tid < o) { s1[tid] += s1[tid + o]; s2[tid] += s2[tid + o]; }
    __syncthreads();
  }
  float mean = s1[0] * (1.0f / C_);
  float var  = s2[0] * (1.0f / C_) - mean * mean;
  float rstd = rsqrtf(var + EPS_);
  __bf16* o = out + (size_t)row * C_;
  o[tid]       = f2bf((x0 - mean) * rstd * g[tid] + be[tid]);
  o[tid + 256] = f2bf((x1 - mean) * rstd * g[tid + 256] + be[tid + 256]);
}

// QKV: block = (b,h,t-tile); 4 waves, wave w owns d-tile w.
// Weights pre-transposed per head: [D x C]. Writes q,k row-major [T,D] and
// V transposed [D,T] (vectorized 16B stores).
__global__ void qkv_kernel(const __bf16* __restrict__ h1, const __bf16* __restrict__ WqT,
                           const __bf16* __restrict__ WkT, const __bf16* __restrict__ WvT,
                           __bf16* __restrict__ qo, __bf16* __restrict__ ko,
                           __bf16* __restrict__ vt) {
  int lane = threadIdx.x & 31, wave = threadIdx.x >> 5;
  int tt = blockIdx.x % (T_ / 16);
  int bh = blockIdx.x / (T_ / 16);
  int b = bh / H_, h = bh % H_;
  int t0 = tt * 16, d0 = wave * 16;
  const __bf16* A  = h1 + (size_t)b * T_ * C_;
  const __bf16* wq = WqT + (size_t)h * D_ * C_;   // [D,C]
  const __bf16* wk = WkT + (size_t)h * D_ * C_;
  const __bf16* wv = WvT + (size_t)h * D_ * C_;
  v8f aq = {}, ak = {}, av = {};
  for (int k0 = 0; k0 < C_; k0 += 32) {
    v16bf a = load_a(A, C_, t0, k0, lane);
    aq = wmma_bf16(a, load_bt(wq, C_, d0, k0, lane), aq);
    ak = wmma_bf16(a, load_bt(wk, C_, d0, k0, lane), ak);
    av = wmma_bf16(a, load_bt(wv, C_, d0, k0, lane), av);
  }
  int n = lane & 15, mh = lane >> 4;
  size_t base = ((size_t)bh * T_ + t0) * D_ + d0 + n;
#pragma unroll
  for (int r0 = 0; r0 < 8; ++r0) {
    size_t idx = base + (size_t)(r0 + 8 * mh) * D_;
    qo[idx] = f2bf(aq[r0]);
    ko[idx] = f2bf(ak[r0]);
  }
  // V^T: row d = d0+n, cols t = t0+8*mh .. +7 : contiguous 16B per lane.
  v8bf vv;
#pragma unroll
  for (int r0 = 0; r0 < 8; ++r0) vv[r0] = f2bf(av[r0]);
  *(v8bf*)(vt + ((size_t)bh * D_ + d0 + n) * T_ + t0 + 8 * mh) = vv;
}

// Pass A: per-KEY-column (max,sum) over the QUERY axis (causal t>=s).
__global__ void attn_stats_kernel(const __bf16* __restrict__ q,
                                  const __bf16* __restrict__ k,
                                  float* __restrict__ cmax, float* __restrict__ csum) {
  int lane = threadIdx.x & 31, wave = threadIdx.x >> 5;
  int bh = blockIdx.x / 32;
  int st = (blockIdx.x % 32) * 4 + wave;          // s-tile index
  const __bf16* qb = q + (size_t)bh * T_ * D_;
  const __bf16* kb = k + (size_t)bh * T_ * D_;
  int s0 = st * 16;
  int n = lane & 15, mh = lane >> 4;
  int sg = s0 + n;
  const float scale = 0.044194173824159216f;      // C^-0.5
  v16bf bk0 = load_bt(kb, D_, s0, 0, lane);       // loop-invariant: hoist
  v16bf bk1 = load_bt(kb, D_, s0, 32, lane);
  float M = -1e30f, S = 0.f;
  for (int tt = st; tt < T_ / 16; ++tt) {
    int t0 = tt * 16;
    v8f acc = {};
    acc = wmma_bf16(load_a(qb, D_, t0, 0, lane), bk0, acc);
    acc = wmma_bf16(load_a(qb, D_, t0, 32, lane), bk1, acc);
    float w[8];
    float tmax = -1e30f;
#pragma unroll
    for (int r0 = 0; r0 < 8; ++r0) {
      int t = t0 + r0 + 8 * mh;
      w[r0] = (t >= sg) ? acc[r0] * scale : -1e30f;
      tmax = fmaxf(tmax, w[r0]);
    }
    tmax = fmaxf(tmax, __shfl_xor(tmax, 16, 32));  // combine m-halves (same col)
    float nM = fmaxf(M, tmax);
    float ss = 0.f;
#pragma unroll
    for (int r0 = 0; r0 < 8; ++r0)
      if (w[r0] > -1e29f) ss += __expf(w[r0] - nM);
    ss += __shfl_xor(ss, 16, 32);
    S = S * __expf(M - nM) + ss;
    M = nM;
  }
  if (lane < 16) {
    cmax[(size_t)bh * T_ + sg] = M;
    csum[(size_t)bh * T_ + sg] = S;
  }
}

// Pass B: 32 query rows (2 t-tiles) per wave so K/V fragments are shared.
// Recompute QK^T tiles, P = exp(w-cmax)/csum, LDS re-layout, P@V^T.
__global__ void attn_out_kernel(const __bf16* __restrict__ q, const __bf16* __restrict__ k,
                                const __bf16* __restrict__ vt,
                                const float* __restrict__ cmax,
                                const float* __restrict__ csum,
                                __bf16* __restrict__ outc) {
  int lane = threadIdx.x;                          // one wave per block
  int pair = blockIdx.x % (T_ / 32);
  int bh = blockIdx.x / (T_ / 32);
  int b = bh / H_, h = bh % H_;
  const __bf16* qb  = q  + (size_t)bh * T_ * D_;
  const __bf16* kb  = k  + (size_t)bh * T_ * D_;
  const __bf16* vtb = vt + (size_t)bh * D_ * T_;   // [D,T]
  int tta = pair * 2, ttb = tta + 1;
  int t0a = tta * 16, t0b = t0a + 16;
  int n = lane & 15, mh = lane >> 4;
  const float scale = 0.044194173824159216f;
  __shared__ alignas(16) __bf16 P[2][16][32];      // per t-tile: 16 t x 32 s
  // Q fragments fixed for this wave.
  v16bf aqa0 = load_a(qb, D_, t0a, 0, lane), aqa1 = load_a(qb, D_, t0a, 32, lane);
  v16bf aqb0 = load_a(qb, D_, t0b, 0, lane), aqb1 = load_a(qb, D_, t0b, 32, lane);
  v8f acc[2][4] = {};                              // [t-tile][d-tile]
  for (int sp = 0; sp <= ttb; sp += 2) {
#pragma unroll
    for (int half = 0; half < 2; ++half) {
      int stile = sp + half;                       // always <= ttb
      int s0 = stile * 16;
      int sg = s0 + n;
      v16bf bk0 = load_bt(kb, D_, s0, 0, lane);
      v16bf bk1 = load_bt(kb, D_, s0, 32, lane);
      float cm = cmax[(size_t)bh * T_ + sg];
      float ri = 1.0f / csum[(size_t)bh * T_ + sg];
      // t-tile B (upper rows): every s-tile in range.
      {
        v8f w = wmma_bf16(aqb0, bk0, (v8f){});
        w = wmma_bf16(aqb1, bk1, w);
#pragma unroll
        for (int r0 = 0; r0 < 8; ++r0) {
          int t = t0b + r0 + 8 * mh;
          float p = (t >= sg) ? __expf(w[r0] * scale - cm) * ri : 0.f;
          P[1][r0 + 8 * mh][n + 16 * half] = f2bf(p);
        }
      }
      // t-tile A (lower rows): only s-tiles <= tta contribute.
      if (stile <= tta) {
        v8f w = wmma_bf16(aqa0, bk0, (v8f){});
        w = wmma_bf16(aqa1, bk1, w);
#pragma unroll
        for (int r0 = 0; r0 < 8; ++r0) {
          int t = t0a + r0 + 8 * mh;
          float p = (t >= sg) ? __expf(w[r0] * scale - cm) * ri : 0.f;
          P[0][r0 + 8 * mh][n + 16 * half] = f2bf(p);
        }
      } else {
#pragma unroll
        for (int r0 = 0; r0 < 8; ++r0) P[0][r0 + 8 * mh][n + 16 * half] = f2bf(0.f);
      }
    }
    // Reload both P tiles as A fragments (two 16B LDS reads each).
    int m = lane & 15, khi = lane >> 4;
    v16bf aP0 = cat8(*(const v8bf*)&P[0][m][8 * khi],
                     *(const v8bf*)&P[0][m][16 + 8 * khi]);
    v16bf aP1 = cat8(*(const v8bf*)&P[1][m][8 * khi],
                     *(const v8bf*)&P[1][m][16 + 8 * khi]);
    int sbase = sp * 16;
#pragma unroll
    for (int dt = 0; dt < 4; ++dt) {
      v16bf vf = load_bt(vtb, T_, dt * 16, sbase, lane);   // shared by both tiles
      acc[0][dt] = wmma_bf16(aP0, vf, acc[0][dt]);
      acc[1][dt] = wmma_bf16(aP1, vf, acc[1][dt]);
    }
  }
#pragma unroll
  for (int ti = 0; ti < 2; ++ti) {
    int t0 = ti ? t0b : t0a;
#pragma unroll
    for (int dt = 0; dt < 4; ++dt)
#pragma unroll
      for (int r0 = 0; r0 < 8; ++r0)
        outc[((size_t)b * T_ + t0 + r0 + 8 * mh) * C_ + h * D_ + dt * 16 + n] =
            f2bf(acc[ti][dt][r0]);
  }
}

// GEMM, 64x32 tile per wave (4x2 WMMA accs), software-pipelined K loop.
// B pre-transposed [N x K].
__global__ void gemm_kernel(const __bf16* __restrict__ A, const __bf16* __restrict__ Bt,
                            const float* __restrict__ bias, const float* __restrict__ resid,
                            float* __restrict__ outF, __bf16* __restrict__ outB,
                            int M, int N, int K, int relu) {
  int lane = threadIdx.x & 31;
  int waveId = blockIdx.x * (blockDim.x >> 5) + (threadIdx.x >> 5);
  int ntx = N >> 5;
  int mt = waveId / ntx, nt = waveId % ntx;
  if (mt >= (M >> 6)) return;                      // wave-uniform guard
  int m0 = mt << 6, n0 = nt << 5;
  v8f acc[4][2] = {};
  v16bf af[4], bf2[2];
#pragma unroll
  for (int i = 0; i < 4; ++i) af[i] = load_a(A, K, m0 + 16 * i, 0, lane);
#pragma unroll
  for (int j = 0; j < 2; ++j) bf2[j] = load_bt(Bt, K, n0 + 16 * j, 0, lane);
  for (int k0 = 32; k0 < K; k0 += 32) {
    if ((k0 & 127) == 32) {                        // one burst per 128 bf16 (256B) of K
      __builtin_prefetch(A + (size_t)(m0 + lane) * K + k0 + 224, 0, 3);
      __builtin_prefetch(A + (size_t)(m0 + 32 + lane) * K + k0 + 224, 0, 3);
      __builtin_prefetch(Bt + (size_t)(n0 + lane) * K + k0 + 224, 0, 3);
    }
    // Issue next k-step's loads before this step's WMMAs (pipeline).
    v16bf naf[4], nbf[2];
#pragma unroll
    for (int i = 0; i < 4; ++i) naf[i] = load_a(A, K, m0 + 16 * i, k0, lane);
#pragma unroll
    for (int j = 0; j < 2; ++j) nbf[j] = load_bt(Bt, K, n0 + 16 * j, k0, lane);
#pragma unroll
    for (int i = 0; i < 4; ++i)
#pragma unroll
      for (int j = 0; j < 2; ++j) acc[i][j] = wmma_bf16(af[i], bf2[j], acc[i][j]);
#pragma unroll
    for (int i = 0; i < 4; ++i) af[i] = naf[i];
#pragma unroll
    for (int j = 0; j < 2; ++j) bf2[j] = nbf[j];
  }
#pragma unroll
  for (int i = 0; i < 4; ++i)
#pragma unroll
    for (int j = 0; j < 2; ++j) acc[i][j] = wmma_bf16(af[i], bf2[j], acc[i][j]);

  int mh = lane >> 4;
#pragma unroll
  for (int i = 0; i < 4; ++i)
#pragma unroll
    for (int j = 0; j < 2; ++j) {
      int n = n0 + j * 16 + (lane & 15);
      float bv = bias ? bias[n] : 0.f;
#pragma unroll
      for (int r0 = 0; r0 < 8; ++r0) {
        int m = m0 + i * 16 + r0 + 8 * mh;
        float val = acc[i][j][r0] + bv;
        if (relu) val = fmaxf(val, 0.f);
        if (resid) val += resid[(size_t)m * N + n];
        if (outF) outF[(size_t)m * N + n] = val;
        if (outB) outB[(size_t)m * N + n] = f2bf(val);
      }
    }
}

// -------------------------------------------------------------------------
extern "C" void kernel_launch(void* const* d_in, const int* in_sizes, int n_in,
                              void* d_out, int out_size, void* d_ws, size_t ws_size,
                              hipStream_t stream) {
  const float* x   = (const float*)d_in[0];
  const float* Wq  = (const float*)d_in[1];
  const float* Wk  = (const float*)d_in[2];
  const float* Wv  = (const float*)d_in[3];
  const float* Wo  = (const float*)d_in[4];
  const float* bo  = (const float*)d_in[5];
  const float* W1  = (const float*)d_in[6];
  const float* b1  = (const float*)d_in[7];
  const float* W2  = (const float*)d_in[8];
  const float* b2  = (const float*)d_in[9];
  const float* g1  = (const float*)d_in[10];
  const float* be1 = (const float*)d_in[11];
  const float* g2  = (const float*)d_in[12];
  const float* be2 = (const float*)d_in[13];

  const size_t MiB = 1024 * 1024;
  char* ws = (char*)d_ws;
  __bf16* h1   = (__bf16*)(ws + 0);              // 8 MiB  [B*T, C]
  __bf16* qb   = (__bf16*)(ws + 8 * MiB);        // 8 MiB  [B,H,T,D]
  __bf16* kb   = (__bf16*)(ws + 16 * MiB);       // 8 MiB  [B,H,T,D]
  __bf16* vt   = (__bf16*)(ws + 24 * MiB);       // 8 MiB  [B,H,D,T] (V^T)
  __bf16* ff1  = (__bf16*)(ws + 0);              // 32 MiB [B*T,4C] reuses h1/q/k/vt
  float*  x2   = (float*)(ws + 32 * MiB);        // 16 MiB [B*T, C]
  __bf16* h2   = (__bf16*)(ws + 48 * MiB);       // 8 MiB
  __bf16* ao   = (__bf16*)(ws + 56 * MiB);       // 8 MiB  attention out [B,T,C]
  float*  cmax = (float*)(ws + 64 * MiB);        // 256 KiB
  float*  csum = (float*)(ws + 64 * MiB + 256 * 1024);
  __bf16* WqT  = (__bf16*)(ws + 65 * MiB);       // 512 KiB each, [H,D,C]
  __bf16* WkT  = (__bf16*)(ws + 65 * MiB + 512 * 1024);
  __bf16* WvT  = (__bf16*)(ws + 66 * MiB);
  __bf16* WoT  = (__bf16*)(ws + 66 * MiB + 512 * 1024);  // [C,C]^T
  __bf16* W1T  = (__bf16*)(ws + 67 * MiB);       // 2 MiB [4C,C]
  __bf16* W2T  = (__bf16*)(ws + 69 * MiB);       // 2 MiB [C,4C]; total 71 MiB

  const int HCD = H_ * C_ * D_;
  const int CC  = C_ * C_;
  const int C4C = C_ * 4 * C_;
  cvt_t_kernel<<<(HCD + 255) / 256, 256, 0, stream>>>(Wq, WqT, C_, D_, HCD);
  cvt_t_kernel<<<(HCD + 255) / 256, 256, 0, stream>>>(Wk, WkT, C_, D_, HCD);
  cvt_t_kernel<<<(HCD + 255) / 256, 256, 0, stream>>>(Wv, WvT, C_, D_, HCD);
  cvt_t_kernel<<<(CC + 255) / 256, 256, 0, stream>>>(Wo, WoT, C_, C_, CC);
  cvt_t_kernel<<<(C4C + 255) / 256, 256, 0, stream>>>(W1, W1T, C_, 4 * C_, C4C);
  cvt_t_kernel<<<(C4C + 255) / 256, 256, 0, stream>>>(W2, W2T, 4 * C_, C_, C4C);

  ln_kernel<<<B_ * T_, 256, 0, stream>>>(x, g1, be1, h1);
  qkv_kernel<<<B_ * H_ * (T_ / 16), 128, 0, stream>>>(h1, WqT, WkT, WvT, qb, kb, vt);
  attn_stats_kernel<<<B_ * H_ * 32, 128, 0, stream>>>(qb, kb, cmax, csum);
  attn_out_kernel<<<B_ * H_ * (T_ / 32), 32, 0, stream>>>(qb, kb, vt, cmax, csum, ao);
  // x2 = x + ao @ Wo + bo         (waves = (M/64)*(N/32), 8 waves/block)
  gemm_kernel<<<(B_ * T_ / 64) * (C_ / 32) / 8, 256, 0, stream>>>(
      ao, WoT, bo, x, x2, nullptr, B_ * T_, C_, C_, 0);
  ln_kernel<<<B_ * T_, 256, 0, stream>>>(x2, g2, be2, h2);
  // ff1 = relu(h2 @ W1 + b1)
  gemm_kernel<<<(B_ * T_ / 64) * (4 * C_ / 32) / 8, 256, 0, stream>>>(
      h2, W1T, b1, nullptr, nullptr, ff1, B_ * T_, 4 * C_, C_, 1);
  // out = x2 + ff1 @ W2 + b2
  gemm_kernel<<<(B_ * T_ / 64) * (C_ / 32) / 8, 256, 0, stream>>>(
      ff1, W2T, b2, x2, (float*)d_out, nullptr, B_ * T_, C_, 4 * C_, 0);

  (void)in_sizes; (void)n_in; (void)out_size; (void)ws_size;
}
